// CausalSelfAttention_3504693313931
// MI455X (gfx1250) — matrix operand order repaired and checked by reading, not compiled, not run on hardware
//
#include <hip/hip_runtime.h>
#include <hip/hip_bf16.h>

// CDNA5 (gfx1250) flash-attention for scores[t,s] = (h[t]@A_h) . h[s], causal
// softmax, O = P @ h.  B=2, T=2048, d=64, H=8.  wave32.
//  - all matmuls: v_wmma_f32_16x16x32_f16 (f32 accum), softmax in f32
//  - fragments load as b128 pairs (contiguous k = 16*half+e runs, A and B alike)
//  - K/V tiles stream memory->LDS via global_load_async_to_lds_b128,
//    double-buffered, synchronized with s_wait_asynccnt (in-order completion)

typedef __attribute__((ext_vector_type(16))) _Float16 v16h;
typedef __attribute__((ext_vector_type(8)))  _Float16 h8;
typedef __attribute__((ext_vector_type(8)))  float    v8f;

#define BB   2
#define TT   2048
#define DD   64
#define HH   8
#define NROW (TT / 16)          // 128 Q row-blocks per (b,h)
#define KV_STRIDE 72            // 32x64 f16 K/V tile rows: 144B (16B aligned)
#define P_STRIDE  40            // 16x32 f16 P tile rows: 80B (16B aligned)

#define NEG_INF (-1e30f)

static __device__ __forceinline__ v8f wmma_f16(v16h a, v16h b, v8f c) {
    // 8 args: (neg_a, A, neg_b, B, c_mod, C, reuse_a, reuse_b)
    return __builtin_amdgcn_wmma_f32_16x16x32_f16(false, a, false, b, (short)0, c,
                                                  false, false);
}

// Async DMA: 16B per lane, memory -> LDS, tracked by ASYNCcnt (ISA ch.10/15.18).
static __device__ __forceinline__ void async_ld_b128(void* lds, const void* gaddr) {
    const unsigned lds_off = (unsigned)(uintptr_t)lds;   // low 32b = LDS address
    asm volatile("global_load_async_to_lds_b128 %0, %1, off"
                 :: "v"(lds_off), "v"((unsigned long long)(uintptr_t)gaddr)
                 : "memory");
}
static __device__ __forceinline__ void wait_async_le8() {
    asm volatile("s_wait_asynccnt 0x8" ::: "memory");
}
static __device__ __forceinline__ void wait_async_0() {
    asm volatile("s_wait_asynccnt 0x0" ::: "memory");
}

// Build a v16h fragment from two contiguous 16B runs (same k order for A and B):
//  A-frag (16x32): lane m=l%16; e0..15 -> K=16*half+e  (two 8-half runs)
//  B-frag (32x16): lane n=l%16; e0..15 -> K=16*half+e  (two 8-half runs)
static __device__ __forceinline__ v16h frag2x8(const _Float16* lo, const _Float16* hi) {
    h8 a = *(const h8*)lo;
    h8 b = *(const h8*)hi;
    v16h r;
#pragma unroll
    for (int i = 0; i < 8; ++i) { r[i] = a[i]; r[8 + i] = b[i]; }
    return r;
}

// ---------------------------------------------------------------- kernel 0
// f32 -> f16 convert; also emit A transposed per head (At[h][e][k] = A[h][k][e])
// so qproj B-fragments are K-contiguous.
__global__ __launch_bounds__(256)
void cvt_f16_kernel(const float* __restrict__ h, const float* __restrict__ A,
                    _Float16* __restrict__ Hf, _Float16* __restrict__ At,
                    int nH, int nA) {
    int i = blockIdx.x * blockDim.x + threadIdx.x;
    if (i < nH) Hf[i] = (_Float16)h[i];
    if (i < nA) {
        const int hh = i / (DD * DD);
        const int k  = (i / DD) % DD;
        const int e  = i % DD;
        At[(size_t)hh * DD * DD + (size_t)e * DD + k] = (_Float16)A[i];
    }
}

// ---------------------------------------------------------------- kernel 1
// Q[bh, t0..t0+15, 0..63] = h_tile(16x64) @ A[h](64x64), via 8 WMMAs/wave.
__global__ __launch_bounds__(32)
void qproj_kernel(const _Float16* __restrict__ Hf, const _Float16* __restrict__ At,
                  _Float16* __restrict__ Qf) {
    const int tile = blockIdx.x;
    const int trow = tile % NROW;
    const int bh   = tile / NROW;
    const int b    = bh / HH;
    const int hh   = bh % HH;
    const int t0   = trow * 16;
    const int l    = threadIdx.x;
    const int half = l >> 4, lm = l & 15;

    const _Float16* hrow = Hf + ((size_t)b * TT + t0 + lm) * DD;   // 128B row
    const v16h ha0 = frag2x8(hrow + 16 * half,      hrow + 16 * half + 8);
    const v16h ha1 = frag2x8(hrow + 32 + 16 * half, hrow + 32 + 16 * half + 8);

    _Float16* qb = Qf + ((size_t)bh * TT + t0) * DD;
#pragma unroll
    for (int nt = 0; nt < 4; ++nt) {
        // B-fragment rows: At[hh][n][k], n fixed per lane, K-contiguous
        const _Float16* arow = At + ((size_t)hh * DD + nt * 16 + lm) * DD;
        const v16h b0 = frag2x8(arow + 16 * half,      arow + 16 * half + 8);
        const v16h b1 = frag2x8(arow + 32 + 16 * half, arow + 32 + 16 * half + 8);
        v8f c = {};
        c = wmma_f16(ha0, b0, c);
        c = wmma_f16(ha1, b1, c);
#pragma unroll
        for (int r = 0; r < 8; ++r)
            qb[(size_t)(r + 8 * half) * DD + nt * 16 + lm] = (_Float16)c[r];
    }
}

// ---------------------------------------------------------------- kernel 2
// Flash attention: one wave owns 16 query rows; iterate 32-key blocks with
// double-buffered async memory->LDS K/V staging.
__global__ __launch_bounds__(32)
void attn_kernel(const _Float16* __restrict__ Hf, const _Float16* __restrict__ Qf,
                 float* __restrict__ out) {
    __shared__ __align__(16) _Float16 kv[2][32][KV_STRIDE]; // double-buffered K/V
    __shared__ __align__(16) _Float16 pl[16][P_STRIDE];     // P transpose staging

    const int tile = blockIdx.x;
    const int trow = tile % NROW;
    const int bh   = tile / NROW;
    const int b    = bh / HH;
    const int hh   = bh % HH;
    const int t0   = trow * 16;
    const int l    = threadIdx.x;
    const int half = l >> 4, lm = l & 15;

    // Q A-fragments (16x64 split into two K=32 fragments), 4 x b128 loads
    const _Float16* qrow = Qf + ((size_t)bh * TT + t0 + lm) * DD;
    const v16h qa0 = frag2x8(qrow + 16 * half,      qrow + 16 * half + 8);
    const v16h qa1 = frag2x8(qrow + 32 + 16 * half, qrow + 32 + 16 * half + 8);

    float M[8], Lr[8];
    v8f O[4];
#pragma unroll
    for (int r = 0; r < 8; ++r) { M[r] = NEG_INF; Lr[r] = 0.f; }
#pragma unroll
    for (int nt = 0; nt < 4; ++nt) O[nt] = (v8f){};

    const _Float16* kbase = Hf + (size_t)b * TT * DD;
    const int nblk = (t0 + 16 + 31) >> 5;        // key blocks covering s <= t0+15

    // Prologue: async-stage key block 0 (lane l owns key row s0+l, 8 x 16B)
    {
        const _Float16* src = kbase + (size_t)l * DD;
#pragma unroll
        for (int c = 0; c < 8; ++c)
            async_ld_b128(&kv[0][l][c * 8], src + c * 8);
    }

    for (int j = 0; j < nblk; ++j) {
        const int s0  = j * 32;
        const int cur = j & 1;
        const bool more = (j + 1 < nblk);
        if (more) {                               // stream next tile behind compute
            const _Float16* src = kbase + (size_t)(s0 + 32 + l) * DD;
#pragma unroll
            for (int c = 0; c < 8; ++c)
                async_ld_b128(&kv[1 - cur][l][c * 8], src + c * 8);
            wait_async_le8();                     // in-order: current 8 are done
        } else {
            wait_async_0();
        }
        __syncthreads();

        // ---- S = Q K^T  (two 16x16 tiles, each 2 WMMAs over d=64)
        // B-frag rows kv[n][k]: K-contiguous -> 2 x ds_load_b128 each
        const _Float16* kr0 = &kv[cur][lm][0];
        const _Float16* kr1 = &kv[cur][16 + lm][0];
        const v16h kb00 = frag2x8(kr0 + 16 * half,      kr0 + 16 * half + 8);
        const v16h kb01 = frag2x8(kr0 + 32 + 16 * half, kr0 + 32 + 16 * half + 8);
        const v16h kb10 = frag2x8(kr1 + 16 * half,      kr1 + 16 * half + 8);
        const v16h kb11 = frag2x8(kr1 + 32 + 16 * half, kr1 + 32 + 16 * half + 8);
        v8f z = {};
        v8f s0t = wmma_f16(qa1, kb01, wmma_f16(qa0, kb00, z));
        v8f s1t = wmma_f16(qa1, kb11, wmma_f16(qa0, kb10, z));

        // ---- causal mask + online softmax (rows live one-per-VGPR across 16 lanes)
#pragma unroll
        for (int r = 0; r < 8; ++r) {
            const int t  = t0 + r + 8 * half;
            const float x0 = ((s0 + lm)      <= t) ? s0t[r] : NEG_INF;
            const float x1 = ((s0 + 16 + lm) <= t) ? s1t[r] : NEG_INF;
            float mx = fmaxf(x0, x1);
#pragma unroll
            for (int off = 8; off; off >>= 1) mx = fmaxf(mx, __shfl_xor(mx, off, 32));
            const float mnew  = fmaxf(M[r], mx);
            const float alpha = __expf(M[r] - mnew);
            const float p0 = __expf(x0 - mnew);
            const float p1 = __expf(x1 - mnew);
            float rs = p0 + p1;
#pragma unroll
            for (int off = 8; off; off >>= 1) rs += __shfl_xor(rs, off, 32);
            Lr[r] = Lr[r] * alpha + rs;
            M[r]  = mnew;
#pragma unroll
            for (int nt = 0; nt < 4; ++nt) O[nt][r] *= alpha;
            pl[r + 8 * half][lm]      = (_Float16)p0;   // stage P for transpose
            pl[r + 8 * half][16 + lm] = (_Float16)p1;
        }
        __syncthreads();

        // ---- O += P(16x32) @ V(32x64): P as A-fragment (2 x b128), V gathered
        const _Float16* pr = &pl[lm][0];
        const v16h pa = frag2x8(pr + 16 * half, pr + 16 * half + 8);
#pragma unroll
        for (int nt = 0; nt < 4; ++nt) {
            v16h vb;                              // kv[k][n]: row-strided gather
#pragma unroll
            for (int e = 0; e < 16; ++e) vb[e] = kv[cur][16 * half + e][nt * 16 + lm];
            O[nt] = wmma_f16(pa, vb, O[nt]);
        }
        // WAR on pl is covered by in-order LDS + the barrier at loop top;
        // kv writes for the next block target the other buffer.
    }

    // ---- epilogue: out[b, t, h, n] = O / L
#pragma unroll
    for (int nt = 0; nt < 4; ++nt)
#pragma unroll
        for (int r = 0; r < 8; ++r) {
            const int t = t0 + r + 8 * half;
            const int n = nt * 16 + lm;
            out[(((size_t)b * TT + t) * HH + hh) * DD + n] = O[nt][r] / Lr[r];
        }
}

// ---------------------------------------------------------------- launch
extern "C" void kernel_launch(void* const* d_in, const int* in_sizes, int n_in,
                              void* d_out, int out_size, void* d_ws, size_t ws_size,
                              hipStream_t stream) {
    (void)in_sizes; (void)n_in; (void)out_size; (void)ws_size;
    const float* h = (const float*)d_in[0];   // [B,T,d]   = 262144 f32
    const float* A = (const float*)d_in[1];   // [H,d,d]   = 32768  f32
    float* out = (float*)d_out;               // [B,T,H*d] = 2097152 f32

    const int nH = BB * TT * DD;              // 262144
    const int nA = HH * DD * DD;              // 32768
    char* ws = (char*)d_ws;
    _Float16* Hf = (_Float16*)ws;                                   // 512 KB
    _Float16* At = (_Float16*)(ws + (size_t)nH * 2);                // 64 KB (A^T)
    _Float16* Qf = (_Float16*)(ws + (size_t)(nH + nA) * 2);         // 4 MB

    cvt_f16_kernel<<<(nH + 255) / 256, 256, 0, stream>>>(h, A, Hf, At, nH, nA);
    qproj_kernel<<<BB * HH * NROW, 32, 0, stream>>>(Hf, At, Qf);
    attn_kernel <<<BB * HH * NROW, 32, 0, stream>>>(Hf, Qf, out);
}
